// SEGNNModel_87909390614539
// MI455X (gfx1250) — matrix operand order repaired and checked by reading, not compiled, not run on hardware
//
#include <hip/hip_runtime.h>
#include <math.h>

// ===========================================================================
// SEGNN forward for MI455X (gfx1250, wave32, WMMA).
//
// All O3 tensor products are lowered to v_wmma_f32_16x16x32_f16 GEMMs:
//   s_out            = [s*a_s | v.a_v]        @ [Wss;Wvs] + b          (K1 x b0)
//   t                =  s                     @  Wsv                   (K2 x b1)
//   v_out_c (c=0..2) = [v_c*a_s | (v x a_v)_c]@ [Wvs2;Wvv] + t*a_v_c   (K3 x b1)
// with silu/sigmoid gating and residuals fused in the epilog.
//
// Weights are pre-swizzled into WMMA-fragment-major f16 so each B fragment is
// one aligned 32B load (2x global_load_b128); A fragments come from LDS as
// 2x ds_load_b128.
//
// d_in pytree layout assumption (jax tree_leaves, dicts sorted by key):
//   0:pos 1:vel 2:charges
//   3.. : params leaves -> embed(6), layers[l]{msg1,msg2,upd1,upd2}(6 each),
//         out1(6), out2(6); each TP: Wss,Wsv,Wvs,Wvs2,Wvv,b
//   117: edge_index [2,E]   118: batch [N]
// ===========================================================================

typedef __attribute__((ext_vector_type(16))) _Float16 v16h;
typedef __attribute__((ext_vector_type(8)))  float    v8f;

#define H0   128
#define H1   64
#define NPG  5
#define Y0C  0.28209479177387814f
#define C1C  0.4886025119029199f
#define EPSC 1e-12f

// compile-time LDS tile maxima (msg1: a0=258,a1=128 -> K1=386->416, K2=258->288, K3=256)
#define K1MAX  416
#define K2MAX  288
#define K3MAX  256
#define B0MAXP 192

struct TPArgs {
  int R, a0, a1, b0, b1;
  int K1P, K2P, K3P, B0P, B1P;
  int mode;        // 0 = msg1 edge-gather, 1 = plain row-major inputs
  int gated;       // silu on s[:b0-b1], sigmoid(s[b0-b1:]) gates on v
  int resid;       // accumulate (+=) into out_s/out_v
  const float* in_s; const float* in_v;            // plain mode [R,a0] / [R,a1*3]
  const float* node_s; const float* node_v;        // msg1 gather sources [N,H0]/[N,H1*3]
  const int* esrc; const int* edst;
  const float* edge_dist; const float* prodq;
  const float* attr_s;                             // nullptr -> constant Y0
  const float* attr_v;                             // [R,3]
  const _Float16* W1;  // fragment-major pad([Wss;Wvs])  [K1P/32][B0P/16][32][16]
  const _Float16* W2;  // fragment-major pad(Wsv)        [K2P/32][B1P/16][32][16]
  const _Float16* W34; // fragment-major pad([Wvs2;Wvv]) [K3P/32][B1P/16][32][16]
  const float* bias;   // [b0]
  float* out_s;        // stride = gated ? b0-b1 : b0
  float* out_v;        // stride = b1*3
};

// ---------------------------------------------------------------------------
// input gathering
// ---------------------------------------------------------------------------
__device__ __forceinline__ float tp_load_s(const TPArgs& A, int r, int k) {
  if (A.mode == 0) {                       // msg1: cat(s[dst], s[src], dist, q)
    if (k < H0)       return A.node_s[(long)A.edst[r] * H0 + k];
    if (k < 2 * H0)   return A.node_s[(long)A.esrc[r] * H0 + (k - H0)];
    if (k == 2 * H0)  return A.edge_dist[r];
    return A.prodq[r];
  }
  return A.in_s[(long)r * A.a0 + k];
}

__device__ __forceinline__ float tp_load_v(const TPArgs& A, int r, int i, int c) {
  if (A.mode == 0) {                       // msg1: cat(v[dst], v[src])
    if (i < H1) return A.node_v[(long)A.edst[r] * (H1 * 3) + i * 3 + c];
    return A.node_v[(long)A.esrc[r] * (H1 * 3) + (i - H1) * 3 + c];
  }
  return A.in_v[(long)r * (A.a1 * 3) + i * 3 + c];
}

// ---------------------------------------------------------------------------
// gfx1250 WMMA fragment layouts (ISA 7.12.2, 16-bit, wave32)
// A 16x32: lane m=lane&15, half=lane>>4; elem j: v=j>>1;
//          k = 2v + (v>=4 ? 8:0) + 8*half + (j&1)
//   -> per lane: two contiguous 8-element k ranges (2x ds_load_b128)
// B 32x16: lane n=lane&15, half=lane>>4; elem j: k = 8*half + (j&7) + 16*(j>>3) (same order as A)
//   -> pre-swizzled fragment-major in memory: one aligned v16h load per lane
// C/D 16x16: vgpr r: M = r + 8*half, N = lane&15
// ---------------------------------------------------------------------------
__device__ __forceinline__ v16h load_a_frag(const _Float16* Ab, int stride, int kk, int lane) {
  int m = lane & 15, half = lane >> 4;
  v16h f;
#pragma unroll
  for (int j = 0; j < 16; ++j) {
    int vv = j >> 1;
    int k = kk + 2 * vv + ((vv >= 4) ? 8 : 0) + 8 * half + (j & 1);
    f[j] = Ab[m * stride + k];
  }
  return f;
}

// ---------------------------------------------------------------------------
// fused tensor-product kernel: 1 block (8 waves) per 16-row tile
// ---------------------------------------------------------------------------
__global__ __launch_bounds__(256) void tp_wmma_kernel(TPArgs A) {
  __shared__ _Float16 sA1[16 * K1MAX];      // [s*a_s | dots]
  __shared__ _Float16 sA2[16 * K2MAX];      // raw s
  __shared__ _Float16 sA3[3][16 * K3MAX];   // [v_c*a_s | cross_c]
  __shared__ float    sSOUT[16 * B0MAXP];
  __shared__ float    sAV[16 * 3];
  __shared__ float    sAS[16];

  const int r0 = blockIdx.x * 16;
  const int tid = threadIdx.x;
  const int lane = tid & 31, wave = tid >> 5;

  // ---- stage attributes ----
  if (tid < 16) {
    int r = r0 + tid;
    float as = Y0C;
    if (r < A.R && A.attr_s) as = A.attr_s[r];
    sAS[tid] = (r < A.R) ? as : 0.f;
  }
  if (tid < 48) {
    int r = r0 + tid / 3;
    sAV[tid] = (r < A.R) ? A.attr_v[(long)r * 3 + (tid % 3)] : 0.f;
  }
  __syncthreads();

  // ---- stage A matrices (f32 gather -> f16 LDS, zero padded) ----
  const int K1 = A.a0 + A.a1;
  for (int idx = tid; idx < 16 * A.K1P; idx += 256) {
    int m = idx / A.K1P, k = idx - m * A.K1P;
    int r = r0 + m; float val = 0.f;
    if (r < A.R && k < K1) {
      if (k < A.a0) val = tp_load_s(A, r, k) * sAS[m];
      else {
        int i = k - A.a0;
        val = tp_load_v(A, r, i, 0) * sAV[m * 3 + 0]
            + tp_load_v(A, r, i, 1) * sAV[m * 3 + 1]
            + tp_load_v(A, r, i, 2) * sAV[m * 3 + 2];
      }
    }
    sA1[m * A.K1P + k] = (_Float16)val;
  }
  for (int idx = tid; idx < 16 * A.K2P; idx += 256) {
    int m = idx / A.K2P, k = idx - m * A.K2P;
    int r = r0 + m; float val = 0.f;
    if (r < A.R && k < A.a0) val = tp_load_s(A, r, k);
    sA2[m * A.K2P + k] = (_Float16)val;
  }
  for (int c = 0; c < 3; ++c) {
    int c1 = (c + 1) % 3, c2 = (c + 2) % 3;
    for (int idx = tid; idx < 16 * A.K3P; idx += 256) {
      int m = idx / A.K3P, k = idx - m * A.K3P;
      int r = r0 + m; float val = 0.f;
      if (r < A.R && k < 2 * A.a1) {
        if (k < A.a1) val = tp_load_v(A, r, k, c) * sAS[m];
        else {
          int i = k - A.a1;      // (v x a_v)_c = v_c1*a_c2 - v_c2*a_c1
          val = tp_load_v(A, r, i, c1) * sAV[m * 3 + c2]
              - tp_load_v(A, r, i, c2) * sAV[m * 3 + c1];
        }
      }
      sA3[c][m * A.K3P + k] = (_Float16)val;
    }
  }
  __syncthreads();

  // ---- phase 1: s_out = A1 @ W1 + bias -> LDS ----
  const int NT0 = A.B0P >> 4;
  for (int t = wave; t < NT0; t += 8) {
    const _Float16* wb = A.W1 + ((long)t * 32 + lane) * 16;
    const long kstride = (long)NT0 * 512;           // f16 elems per k-step
    v8f acc = {};
    for (int kk = 0; kk < A.K1P; kk += 32) {
      v16h a = load_a_frag(sA1, A.K1P, kk, lane);
      v16h b = *(const v16h*)(wb + (long)(kk >> 5) * kstride);
      acc = __builtin_amdgcn_wmma_f32_16x16x32_f16(false, a, false, b,
                                                   (short)0, acc, false, false);
    }
    int n = t * 16 + (lane & 15);
    float bn = (A.bias && n < A.b0) ? A.bias[n] : 0.f;   // hoisted, loaded once
    int half8 = 8 * (lane >> 4);
#pragma unroll
    for (int rr = 0; rr < 8; ++rr)
      sSOUT[(rr + half8) * B0MAXP + n] = acc[rr] + bn;
  }
  __syncthreads();

  // ---- write scalar outputs (silu if gated) ----
  const int b0eff = A.gated ? (A.b0 - A.b1) : A.b0;
  if (A.out_s) {
    for (int idx = tid; idx < 16 * b0eff; idx += 256) {
      int m = idx / b0eff, n = idx - m * b0eff;
      int r = r0 + m;
      if (r < A.R) {
        float x = sSOUT[m * B0MAXP + n];
        if (A.gated) x = x / (1.f + __expf(-x));          // silu
        long o = (long)r * b0eff + n;
        if (A.resid) A.out_s[o] += x; else A.out_s[o] = x;
      }
    }
  }

  // ---- phase 2: v_out tiles (t = A2@W2, 3x A3_c@W34, recombine + gate) ----
  const int NT1 = A.B1P >> 4;
  for (int t = wave; t < NT1; t += 8) {
    const long kstride1 = (long)NT1 * 512;
    const _Float16* wb2 = A.W2 + ((long)t * 32 + lane) * 16;
    v8f accT = {};
    for (int kk = 0; kk < A.K2P; kk += 32) {
      v16h a = load_a_frag(sA2, A.K2P, kk, lane);
      v16h b = *(const v16h*)(wb2 + (long)(kk >> 5) * kstride1);
      accT = __builtin_amdgcn_wmma_f32_16x16x32_f16(false, a, false, b,
                                                    (short)0, accT, false, false);
    }
    v8f accC[3];
    const _Float16* wb34 = A.W34 + ((long)t * 32 + lane) * 16;
    for (int c = 0; c < 3; ++c) {
      v8f acc = {};
      for (int kk = 0; kk < A.K3P; kk += 32) {
        v16h a = load_a_frag(&sA3[c][0], A.K3P, kk, lane);
        v16h b = *(const v16h*)(wb34 + (long)(kk >> 5) * kstride1);
        acc = __builtin_amdgcn_wmma_f32_16x16x32_f16(false, a, false, b,
                                                     (short)0, acc, false, false);
      }
      accC[c] = acc;
    }
    int n = t * 16 + (lane & 15);
    int half8 = 8 * (lane >> 4);
    if (n < A.b1) {
#pragma unroll
      for (int rr = 0; rr < 8; ++rr) {
        int m = rr + half8;
        int r = r0 + m;
        if (r < A.R) {
          float gate = 1.f;
          if (A.gated) {
            float g = sSOUT[m * B0MAXP + (A.b0 - A.b1) + n];
            gate = 1.f / (1.f + __expf(-g));
          }
          float av0 = sAV[m * 3 + 0], av1 = sAV[m * 3 + 1], av2 = sAV[m * 3 + 2];
          float o0 = (accC[0][rr] + accT[rr] * av0) * gate;
          float o1 = (accC[1][rr] + accT[rr] * av1) * gate;
          float o2 = (accC[2][rr] + accT[rr] * av2) * gate;
          float* dst = A.out_v + (long)r * (A.b1 * 3) + n * 3;
          if (A.resid) { o0 += dst[0]; o1 += dst[1]; o2 += dst[2]; }
          dst[0] = o0; dst[1] = o1; dst[2] = o2;
        }
      }
    }
  }
}

// ---------------------------------------------------------------------------
// weight packing: f32 [Wa;Wb] (concat along K, zero padded) -> f16 in
// WMMA-fragment-major order: out[((kt*NT + t)*32 + lane)*16 + j]
//   with n = t*16 + (lane&15), k = kt*32 + 8*(lane>>4) + (j&7) + ((j>>3)<<4)
// ---------------------------------------------------------------------------
__global__ void pack_frag_f16(const float* Wa, int ka, const float* Wb, int kb,
                              int ncols, int KP, int NP, _Float16* out) {
  int idx = blockIdx.x * blockDim.x + threadIdx.x;
  if (idx >= KP * NP) return;
  int j    = idx & 15;
  int lane = (idx >> 4) & 31;
  int ft   = idx >> 9;                 // kt*NT + t
  int NT   = NP >> 4;
  int kt   = ft / NT, t = ft - kt * NT;
  int n = t * 16 + (lane & 15);
  int k = kt * 32 + 8 * (lane >> 4) + (j & 7) + ((j >> 3) << 4);
  float v = 0.f;
  if (n < ncols) {
    if (k < ka)                 v = Wa[(long)k * ncols + n];
    else if (Wb && k - ka < kb) v = Wb[(long)(k - ka) * ncols + n];
  }
  out[idx] = (_Float16)v;
}

// ---------------------------------------------------------------------------
// geometry / attribute kernels
// ---------------------------------------------------------------------------
__global__ void edge_geom_kernel(const float* pos, const float* charges,
                                 const int* esrc, const int* edst,
                                 float* edist, float* prodq, float* eav, int E) {
  int e = blockIdx.x * blockDim.x + threadIdx.x;
  if (e >= E) return;
  int s = esrc[e], d = edst[e];
  float rx = pos[s * 3 + 0] - pos[d * 3 + 0];
  float ry = pos[s * 3 + 1] - pos[d * 3 + 1];
  float rz = pos[s * 3 + 2] - pos[d * 3 + 2];
  float dist = sqrtf(rx * rx + ry * ry + rz * rz);
  edist[e] = dist;
  prodq[e] = charges[s] * charges[d];
  float inv = C1C / (dist + EPSC);
  eav[e * 3 + 0] = rx * inv; eav[e * 3 + 1] = ry * inv; eav[e * 3 + 2] = rz * inv;
}

__global__ void node_geom_kernel(const float* pos, const float* vel, const int* batch,
                                 const int* edst, const float* eav,
                                 float* velabs, float* cpos, float* nas, float* nav,
                                 int N, int EPG) {
  int n = blockIdx.x * blockDim.x + threadIdx.x;
  if (n >= N) return;
  float vx = vel[n * 3], vy = vel[n * 3 + 1], vz = vel[n * 3 + 2];
  float va = sqrtf(vx * vx + vy * vy + vz * vz);
  velabs[n] = va;
  float vi = C1C / (va + EPSC);
  int g = batch[n];
  // deterministic scatter-mean of edge attrs onto this receiver
  float deg = 0.f, sas = 0.f, s0 = 0.f, s1 = 0.f, s2 = 0.f;
  int eb = g * EPG;
  for (int j = 0; j < EPG; ++j) {
    int e = eb + j;
    if (edst[e] == n) {
      deg += 1.f; sas += Y0C;
      s0 += eav[e * 3]; s1 += eav[e * 3 + 1]; s2 += eav[e * 3 + 2];
    }
  }
  float di = 1.f / fmaxf(deg, 1.f);
  nas[n] = sas * di + Y0C;
  nav[n * 3 + 0] = s0 * di + vx * vi;
  nav[n * 3 + 1] = s1 * di + vy * vi;
  nav[n * 3 + 2] = s2 * di + vz * vi;
  int nb = g * NPG;
  float mx = 0.f, my = 0.f, mz = 0.f;
  for (int j = 0; j < NPG; ++j) {
    mx += pos[(nb + j) * 3]; my += pos[(nb + j) * 3 + 1]; mz += pos[(nb + j) * 3 + 2];
  }
  cpos[n * 3 + 0] = pos[n * 3 + 0] - mx / NPG;
  cpos[n * 3 + 1] = pos[n * 3 + 1] - my / NPG;
  cpos[n * 3 + 2] = pos[n * 3 + 2] - mz / NPG;
}

// embed: a0=1 (vel_abs), a1=2 (cpos, vel) -> b0=128, b1=64 (tiny K: VALU)
__global__ void embed_kernel(const float* velabs, const float* cpos, const float* vel,
                             const float* nas, const float* nav,
                             const float* Wss, const float* Wsv, const float* Wvs,
                             const float* Wvs2, const float* Wvv, const float* bias,
                             float* s_out, float* v_out, int N) {
  int idx = blockIdx.x * blockDim.x + threadIdx.x;
  int n = idx >> 7, f = idx & 127;
  if (n >= N) return;
  float as = nas[n];
  float a0 = nav[n * 3], a1 = nav[n * 3 + 1], a2 = nav[n * 3 + 2];
  float s0 = velabs[n];
  float p0 = cpos[n * 3], p1 = cpos[n * 3 + 1], p2 = cpos[n * 3 + 2];
  float w0 = vel[n * 3],  w1 = vel[n * 3 + 1],  w2 = vel[n * 3 + 2];
  float dot0 = p0 * a0 + p1 * a1 + p2 * a2;
  float dot1 = w0 * a0 + w1 * a1 + w2 * a2;
  s_out[(long)n * H0 + f] = s0 * as * Wss[f] + dot0 * Wvs[f] + dot1 * Wvs[H0 + f] + bias[f];
  if (f < H1) {
    float t = s0 * Wsv[f];
    float c00 = p1 * a2 - p2 * a1, c01 = p2 * a0 - p0 * a2, c02 = p0 * a1 - p1 * a0;
    float c10 = w1 * a2 - w2 * a1, c11 = w2 * a0 - w0 * a2, c12 = w0 * a1 - w1 * a0;
    float u2a = Wvs2[f], u2b = Wvs2[H1 + f];
    float uva = Wvv[f],  uvb = Wvv[H1 + f];
    long base = (long)n * (H1 * 3) + f * 3;
    v_out[base + 0] = t * a0 + p0 * as * u2a + w0 * as * u2b + c00 * uva + c10 * uvb;
    v_out[base + 1] = t * a1 + p1 * as * u2a + w1 * as * u2b + c01 * uva + c11 * uvb;
    v_out[base + 2] = t * a2 + p2 * as * u2a + w2 * as * u2b + c02 * uva + c12 * uvb;
  }
}

// deterministic message aggregation (aggr='add') by scanning the graph's edges
__global__ void aggregate_kernel(const float* es, const float* ev,
                                 const int* batch, const int* edst,
                                 float* aggs, float* aggv, int N, int EPG) {
  int idx = blockIdx.x * blockDim.x + threadIdx.x;
  int n = idx / 320, f = idx - n * 320;    // 128 scalars + 64*3 vector comps
  if (n >= N) return;
  int g = batch[n], eb = g * EPG;
  float acc = 0.f;
  for (int j = 0; j < EPG; ++j) {
    int e = eb + j;
    if (edst[e] == n)
      acc += (f < H0) ? es[(long)e * H0 + f] : ev[(long)e * (H1 * 3) + (f - H0)];
  }
  if (f < H0) aggs[(long)n * H0 + f] = acc;
  else        aggv[(long)n * (H1 * 3) + (f - H0)] = acc;
}

__global__ void concat_kernel(const float* s, const float* as_, const float* v, const float* av_,
                              float* cs, float* cv, int N) {
  int idx = blockIdx.x * blockDim.x + threadIdx.x;
  int n = idx / 640, f = idx - n * 640;
  if (n >= N) return;
  if (f < 256) cs[(long)n * 256 + f] = (f < H0) ? s[(long)n * H0 + f] : as_[(long)n * H0 + (f - H0)];
  else {
    int j = f - 256;
    cv[(long)n * 384 + j] = (j < 192) ? v[(long)n * 192 + j] : av_[(long)n * 192 + (j - 192)];
  }
}

// out2: a0=128, a1=64 -> 1 x 1o per node (Nout=1: VALU)
__global__ void out2_kernel(const float* s, const float* v, const float* nas, const float* nav,
                            const float* Wsv, const float* Wvs2, const float* Wvv,
                            const float* pos, float* out, int N) {
  int n = blockIdx.x * blockDim.x + threadIdx.x;
  if (n >= N) return;
  float as = nas[n];
  float a0 = nav[n * 3], a1 = nav[n * 3 + 1], a2 = nav[n * 3 + 2];
  float t = 0.f;
  for (int k = 0; k < H0; ++k) t += s[(long)n * H0 + k] * Wsv[k];
  float o0 = t * a0, o1 = t * a1, o2 = t * a2;
  for (int i = 0; i < H1; ++i) {
    float x = v[(long)n * 192 + i * 3], y = v[(long)n * 192 + i * 3 + 1], z = v[(long)n * 192 + i * 3 + 2];
    float w2 = Wvs2[i], wv = Wvv[i];
    o0 += x * as * w2 + (y * a2 - z * a1) * wv;
    o1 += y * as * w2 + (z * a0 - x * a2) * wv;
    o2 += z * as * w2 + (x * a1 - y * a0) * wv;
  }
  out[n * 3 + 0] = pos[n * 3 + 0] + o0;
  out[n * 3 + 1] = pos[n * 3 + 1] + o1;
  out[n * 3 + 2] = pos[n * 3 + 2] + o2;
}

// ===========================================================================
// host orchestration
// ===========================================================================
struct TPW { const float *Wss, *Wsv, *Wvs, *Wvs2, *Wvv, *b; };
struct Dims { int a0, a1, b0, b1, K1P, K2P, K3P, B0P, B1P; };
struct PackedW { _Float16 *W1, *W2, *W34; };

static inline int rnd32(int x) { return (x + 31) & ~31; }
static inline int rnd16(int x) { return (x + 15) & ~15; }
static inline Dims mkdims(int a0, int a1, int b0, int b1) {
  Dims d; d.a0 = a0; d.a1 = a1; d.b0 = b0; d.b1 = b1;
  d.K1P = rnd32(a0 + a1); d.K2P = rnd32(a0); d.K3P = rnd32(2 * a1);
  d.B0P = rnd16(b0); d.B1P = rnd16(b1);
  return d;
}

extern "C" void kernel_launch(void* const* d_in, const int* in_sizes, int n_in,
                              void* d_out, int out_size, void* d_ws, size_t ws_size,
                              hipStream_t stream) {
  (void)n_in; (void)out_size; (void)ws_size;
  const float* pos     = (const float*)d_in[0];
  const float* vel     = (const float*)d_in[1];
  const float* charges = (const float*)d_in[2];
  const int PB = 3;
  auto get_tp = [&](int base) -> TPW {
    TPW w;
    w.Wss  = (const float*)d_in[base + 0];
    w.Wsv  = (const float*)d_in[base + 1];
    w.Wvs  = (const float*)d_in[base + 2];
    w.Wvs2 = (const float*)d_in[base + 3];
    w.Wvv  = (const float*)d_in[base + 4];
    w.b    = (const float*)d_in[base + 5];
    return w;
  };
  // params: embed(6) | layers 4x{msg1,msg2,upd1,upd2}(24) | out1(6) | out2(6)
  TPW w_embed = get_tp(PB);
  TPW w_layer[4][4];
  for (int l = 0; l < 4; ++l)
    for (int t = 0; t < 4; ++t)
      w_layer[l][t] = get_tp(PB + 6 + (l * 4 + t) * 6);
  TPW w_out1 = get_tp(PB + 6 + 96);
  TPW w_out2 = get_tp(PB + 6 + 96 + 6);
  const int* eidx  = (const int*)d_in[PB + 114];
  const int* batch = (const int*)d_in[PB + 115];

  const int N = in_sizes[0] / 3;
  const int E = in_sizes[PB + 114] / 2;
  const int* esrc = eidx;
  const int* edst = eidx + E;
  const int EPG = (N > 0) ? (E * NPG) / N : 20;   // edges per graph (20)

  // ---- workspace bump allocator ----
  char* wp = (char*)d_ws;
  auto alloc = [&](size_t bytes) -> void* {
    void* r = (void*)wp; wp += (bytes + 255) & ~(size_t)255; return r;
  };
  float* cpos   = (float*)alloc((size_t)N * 3 * 4);
  float* velabs = (float*)alloc((size_t)N * 4);
  float* nas    = (float*)alloc((size_t)N * 4);
  float* nav    = (float*)alloc((size_t)N * 3 * 4);
  float* edist  = (float*)alloc((size_t)E * 4);
  float* prodq  = (float*)alloc((size_t)E * 4);
  float* eav    = (float*)alloc((size_t)E * 3 * 4);
  float* s_node = (float*)alloc((size_t)N * H0 * 4);
  float* v_node = (float*)alloc((size_t)N * H1 * 3 * 4);
  float* e1s    = (float*)alloc((size_t)E * H0 * 4);
  float* e1v    = (float*)alloc((size_t)E * H1 * 3 * 4);
  float* e2s    = (float*)alloc((size_t)E * H0 * 4);
  float* e2v    = (float*)alloc((size_t)E * H1 * 3 * 4);
  float* aggs   = (float*)alloc((size_t)N * H0 * 4);
  float* aggv   = (float*)alloc((size_t)N * H1 * 3 * 4);
  float* cats   = (float*)alloc((size_t)N * 256 * 4);
  float* catv   = (float*)alloc((size_t)N * 384 * 4);
  float* us     = (float*)alloc((size_t)N * H0 * 4);
  float* uv     = (float*)alloc((size_t)N * H1 * 3 * 4);

  // ---- pack weights to fragment-major f16 ----
  const Dims Dmsg1 = mkdims(2 * H0 + 2, 2 * H1, H0 + H1, H1);
  const Dims Dmsg2 = mkdims(H0, H1, H0 + H1, H1);
  const Dims Dupd1 = mkdims(2 * H0, 2 * H1, H0 + H1, H1);
  const Dims Dupd2 = mkdims(H0, H1, H0, H1);
  const Dims Dout1 = mkdims(H0, H1, H0 + H1, H1);

  auto pack_tp = [&](const TPW& w, const Dims& D) -> PackedW {
    PackedW P;
    P.W1  = (_Float16*)alloc((size_t)D.K1P * D.B0P * 2);
    P.W2  = (_Float16*)alloc((size_t)D.K2P * D.B1P * 2);
    P.W34 = (_Float16*)alloc((size_t)D.K3P * D.B1P * 2);
    int n1 = D.K1P * D.B0P, n2 = D.K2P * D.B1P, n3 = D.K3P * D.B1P;
    pack_frag_f16<<<(n1 + 255) / 256, 256, 0, stream>>>(w.Wss, D.a0, w.Wvs, D.a1, D.b0, D.K1P, D.B0P, P.W1);
    pack_frag_f16<<<(n2 + 255) / 256, 256, 0, stream>>>(w.Wsv, D.a0, nullptr, 0, D.b1, D.K2P, D.B1P, P.W2);
    pack_frag_f16<<<(n3 + 255) / 256, 256, 0, stream>>>(w.Wvs2, D.a1, w.Wvv, D.a1, D.b1, D.K3P, D.B1P, P.W34);
    return P;
  };
  PackedW p_layer[4][4];
  for (int l = 0; l < 4; ++l) {
    p_layer[l][0] = pack_tp(w_layer[l][0], Dmsg1);
    p_layer[l][1] = pack_tp(w_layer[l][1], Dmsg2);
    p_layer[l][2] = pack_tp(w_layer[l][2], Dupd1);
    p_layer[l][3] = pack_tp(w_layer[l][3], Dupd2);
  }
  PackedW p_out1 = pack_tp(w_out1, Dout1);

  // ---- geometry + embedding ----
  edge_geom_kernel<<<(E + 255) / 256, 256, 0, stream>>>(pos, charges, esrc, edst,
                                                        edist, prodq, eav, E);
  node_geom_kernel<<<(N + 255) / 256, 256, 0, stream>>>(pos, vel, batch, edst, eav,
                                                        velabs, cpos, nas, nav, N, EPG);
  {
    long tot = (long)N * 128;
    embed_kernel<<<(tot + 255) / 256, 256, 0, stream>>>(velabs, cpos, vel, nas, nav,
        w_embed.Wss, w_embed.Wsv, w_embed.Wvs, w_embed.Wvs2, w_embed.Wvv, w_embed.b,
        s_node, v_node, N);
  }

  // ---- generic TP launcher ----
  auto launch_tp = [&](int R, const Dims& D, const PackedW& P, const float* bias,
                       int mode, const float* ins, const float* inv,
                       const float* attr_s, const float* attr_v,
                       int gated, int resid, float* outs, float* outv) {
    TPArgs A;
    A.R = R; A.a0 = D.a0; A.a1 = D.a1; A.b0 = D.b0; A.b1 = D.b1;
    A.K1P = D.K1P; A.K2P = D.K2P; A.K3P = D.K3P; A.B0P = D.B0P; A.B1P = D.B1P;
    A.mode = mode; A.gated = gated; A.resid = resid;
    A.in_s = ins; A.in_v = inv;
    A.node_s = s_node; A.node_v = v_node;
    A.esrc = esrc; A.edst = edst; A.edge_dist = edist; A.prodq = prodq;
    A.attr_s = attr_s; A.attr_v = attr_v;
    A.W1 = P.W1; A.W2 = P.W2; A.W34 = P.W34; A.bias = bias;
    A.out_s = outs; A.out_v = outv;
    tp_wmma_kernel<<<(R + 15) / 16, 256, 0, stream>>>(A);
  };

  // ---- message passing layers ----
  for (int l = 0; l < 4; ++l) {
    // msg1: edge gather of node features, edge attrs (a_s = Y0), gated
    launch_tp(E, Dmsg1, p_layer[l][0], w_layer[l][0].b,
              /*mode*/0, nullptr, nullptr, /*attr_s*/nullptr, eav,
              /*gated*/1, /*resid*/0, e1s, e1v);
    // msg2: plain edge inputs, gated
    launch_tp(E, Dmsg2, p_layer[l][1], w_layer[l][1].b,
              1, e1s, e1v, nullptr, eav, 1, 0, e2s, e2v);
    // aggregate messages onto receivers (deterministic scan)
    {
      long tot = (long)N * 320;
      aggregate_kernel<<<(tot + 255) / 256, 256, 0, stream>>>(e2s, e2v, batch, edst,
                                                              aggs, aggv, N, EPG);
    }
    // concat node state with aggregated message
    {
      long tot = (long)N * 640;
      concat_kernel<<<(tot + 255) / 256, 256, 0, stream>>>(s_node, aggs, v_node, aggv,
                                                           cats, catv, N);
    }
    // upd1: gated, node attrs
    launch_tp(N, Dupd1, p_layer[l][2], w_layer[l][2].b,
              1, cats, catv, nas, nav, 1, 0, us, uv);
    // upd2: plain TP, residual add into node state
    launch_tp(N, Dupd2, p_layer[l][3], w_layer[l][3].b,
              1, us, uv, nas, nav, 0, 1, s_node, v_node);
  }

  // ---- output head ----
  launch_tp(N, Dout1, p_out1, w_out1.b, 1, s_node, v_node, nas, nav, 1, 0, us, uv);
  out2_kernel<<<(N + 255) / 256, 256, 0, stream>>>(us, uv, nas, nav,
      w_out2.Wsv, w_out2.Wvs2, w_out2.Wvv, pos, (float*)d_out, N);
}
